// MultiHeadAttention_6897717477658
// MI455X (gfx1250) — matrix operand order repaired and checked by reading, not compiled, not run on hardware
//
#include <hip/hip_runtime.h>

#define B_    8
#define S_    16
#define INDIM 4096
#define H_    32
#define D_    128
#define PAST_ 4096
#define KV_   (PAST_ + S_)   // 4112

typedef float        v4f   __attribute__((ext_vector_type(4)));
typedef float        v8f   __attribute__((ext_vector_type(8)));
typedef __bf16       v16bf __attribute__((ext_vector_type(16)));
typedef unsigned int v4u   __attribute__((ext_vector_type(4)));

union Frag16 {
  v16bf  v;
  __bf16 h[16];
  v4u    q[2];
};

// ---------------------------------------------------------------------------
// CDNA5 inline-asm helpers: async global->LDS + split wait counters
// ---------------------------------------------------------------------------
__device__ __forceinline__ void async_load_b128(unsigned lds_addr, unsigned long long gaddr) {
  asm volatile("global_load_async_to_lds_b128 %0, %1, off"
               :: "v"(lds_addr), "v"(gaddr) : "memory");
}
__device__ __forceinline__ void wait_async0() {
  asm volatile("s_wait_asynccnt 0" ::: "memory");
}
__device__ __forceinline__ void wait_ds0() {
  asm volatile("s_wait_dscnt 0" ::: "memory");
}

// ---------------------------------------------------------------------------
// bf16 fragment builders, contiguous k = 16*(lane/16) + s in both A and B (wave32)
// ---------------------------------------------------------------------------
__device__ __forceinline__ v16bf pack16(const float* f) {
  Frag16 r;
#pragma unroll
  for (int i = 0; i < 16; ++i) r.h[i] = (__bf16)f[i];
  return r.v;
}

// A-matrix 16x32 bf16: lane group g holds K = 16g..16g+15 of row (lane&15)
__device__ __forceinline__ v16bf load_a_frag(const float* __restrict__ row, int k, int g) {
  const v4f* p0 = (const v4f*)(row + k + 16 * g);
  const v4f* p1 = (const v4f*)(row + k + 16 * g + 8);
  v4f a = p0[0], b = p0[1], c = p1[0], d = p1[1];
  float t[16] = {a.x, a.y, a.z, a.w, b.x, b.y, b.z, b.w,
                 c.x, c.y, c.z, c.w, d.x, d.y, d.z, d.w};
  return pack16(t);
}

// B-matrix 32x16 bf16: lane holds column (lane&15), K = 16g..16g+15 -> 16 contiguous floats
__device__ __forceinline__ v16bf load_b_frag16(const float* __restrict__ p) {
  const v4f* q = (const v4f*)p;
  v4f a = q[0], b = q[1], c = q[2], d = q[3];
  float t[16] = {a.x, a.y, a.z, a.w, b.x, b.y, b.z, b.w,
                 c.x, c.y, c.z, c.w, d.x, d.y, d.z, d.w};
  return pack16(t);
}

__device__ __forceinline__ v8f wmma_bf16(v16bf a, v16bf b, v8f c) {
  return __builtin_amdgcn_wmma_f32_16x16x32_bf16(false, a, false, b, (short)0, c,
                                                 false, false);
}

// ---------------------------------------------------------------------------
// GEMM: Y[m,n] = sum_k X[m,k] * W[n,k]   (M=128, N=4096, K=4096)
// one wave computes a 16(M) x 64(N) tile; mode selects output layout
//   mode 0: Q   -> ws   [B,H,S,D]
//   mode 1/2: K/V new -> d_out concat slot [B,H,KV,D] rows PAST..PAST+15
//   mode 3: final out -> d_out [B,S,4096]
// ---------------------------------------------------------------------------
__global__ __launch_bounds__(32)
void proj_gemm(const float* __restrict__ X, const float* __restrict__ W,
               float* __restrict__ Y, int mode) {
  const int lane = threadIdx.x & 31;
  const int g = lane >> 4, ln = lane & 15;
  const int ntile = blockIdx.x * 64;
  const int b = blockIdx.y;  // 16 M-rows per tile == one batch (S=16)

  const float* xrow = X + (size_t)(b * 16 + ln) * INDIM;

  v8f acc[4];
#pragma unroll
  for (int j = 0; j < 4; ++j)
#pragma unroll
    for (int r = 0; r < 8; ++r) acc[j][r] = 0.0f;

  for (int k = 0; k < INDIM; k += 32) {
    v16bf af = load_a_frag(xrow, k, g);
#pragma unroll
    for (int j = 0; j < 4; ++j) {
      const float* wrow = W + (size_t)(ntile + j * 16 + ln) * INDIM + k + 16 * g;
      v16bf bf = load_b_frag16(wrow);
      acc[j] = wmma_bf16(af, bf, acc[j]);
    }
  }

#pragma unroll
  for (int j = 0; j < 4; ++j) {
#pragma unroll
    for (int r = 0; r < 8; ++r) {
      const int s = r + 8 * g;                 // D-layout: row = vgpr + 8*(lane/16)
      const int n = ntile + j * 16 + ln;       //           col = lane&15
      const float v = acc[j][r];
      if (mode == 0) {
        const int h = n >> 7, d = n & 127;
        Y[(((size_t)b * H_ + h) * S_ + s) * D_ + d] = v;
      } else if (mode <= 2) {
        const int h = n >> 7, d = n & 127;
        Y[((size_t)(b * H_ + h) * KV_ + PAST_ + s) * D_ + d] = v;
      } else {
        Y[(size_t)(b * 16 + s) * INDIM + n] = v;
      }
    }
  }
}

// ---------------------------------------------------------------------------
// KV cache concat copy: pure streaming, B128 non-temporal
// ---------------------------------------------------------------------------
__global__ void copy_cache(const float* __restrict__ src, float* __restrict__ dst,
                           long nvec) {
  const long CHUNK = (long)PAST_ * D_ / 4;  // 131072 vec4 per (b,h)
  const long DSTCH = (long)KV_ * D_ / 4;    // 131584 vec4 per (b,h)
  for (long i = blockIdx.x * (long)blockDim.x + threadIdx.x; i < nvec;
       i += (long)gridDim.x * blockDim.x) {
    const long bh = i >> 17;                 // / CHUNK (2^17)
    const long rem = i - bh * CHUNK;
    const v4f* s = (const v4f*)src + i;
    v4f* d = (v4f*)dst + bh * DSTCH + rem;
    __builtin_nontemporal_store(__builtin_nontemporal_load(s), d);
  }
}

// ---------------------------------------------------------------------------
// Flash attention: one wave per (b,h). S=16 -> one M tile, D=128, KV chunked by 32.
// V staged global->LDS with async tensor-style copies (ASYNCcnt), P staged
// through LDS (bf16) to convert D-layout -> A-layout for the PV WMMA.
// ---------------------------------------------------------------------------
__global__ __launch_bounds__(32)
void flash_attn(const float* __restrict__ Q, const float* __restrict__ K,
                const float* __restrict__ V, const int* __restrict__ mask,
                float* __restrict__ O) {
  __shared__ __align__(16) float  sV[32 * 132];    // V rows, padded stride (16B mult)
  __shared__ __align__(16) __bf16 sP[16 * 32];     // P tile, row stride 32

  const int lane = threadIdx.x & 31;
  const int g = lane >> 4, ln = lane & 15;
  const int bh = blockIdx.x, b = bh >> 5, h = bh & 31;
  const float scale = 0.08838834764831845f;  // 1/sqrt(128)

  // preload Q fragments for all 4 K-steps (D=128)
  const float* qrow = Q + ((size_t)bh * S_ + ln) * D_;
  v16bf qf[4];
#pragma unroll
  for (int st = 0; st < 4; ++st) qf[st] = load_a_frag(qrow, st * 32, g);

  v8f acc[8];
  float m_run[8], l_run[8];
#pragma unroll
  for (int j = 0; j < 8; ++j)
#pragma unroll
    for (int r = 0; r < 8; ++r) acc[j][r] = 0.0f;
#pragma unroll
  for (int r = 0; r < 8; ++r) { m_run[r] = -1e30f; l_run[r] = 0.0f; }

  const float* Kb = K + (size_t)bh * KV_ * D_;
  const float* Vb = V + (size_t)bh * KV_ * D_;
  const unsigned ldsV = (unsigned)(unsigned long long)&sV[0];
  const unsigned myLds = ldsV + (unsigned)lane * (132 * 4);

  for (int kv0 = 0; kv0 < KV_; kv0 += 32) {
    // WAR fence: previous chunk's DS reads of sV must drain before async rewrites
    wait_ds0();

    // async-stage this chunk's V rows: lane owns one row (clamped at tail)
    {
      int kv = kv0 + lane;
      if (kv > KV_ - 1) kv = KV_ - 1;
      const unsigned long long ga = (unsigned long long)(Vb + (size_t)kv * D_);
#pragma unroll
      for (int d4 = 0; d4 < 32; ++d4)
        async_load_b128(myLds + d4 * 16, ga + (unsigned long long)d4 * 16);
    }

    // logits: two 16x16 tiles (kv0..+15, kv0+16..+31), K-dim = D = 4 steps of 32
    v8f c0, c1;
#pragma unroll
    for (int r = 0; r < 8; ++r) { c0[r] = 0.0f; c1[r] = 0.0f; }
    int kr0 = kv0 + ln;       if (kr0 > KV_ - 1) kr0 = KV_ - 1;
    int kr1 = kv0 + 16 + ln;  if (kr1 > KV_ - 1) kr1 = KV_ - 1;
    const float* k0 = Kb + (size_t)kr0 * D_;
    const float* k1 = Kb + (size_t)kr1 * D_;
#pragma unroll
    for (int st = 0; st < 4; ++st) {
      v16bf b0 = load_b_frag16(k0 + st * 32 + 16 * g);
      v16bf b1 = load_b_frag16(k1 + st * 32 + 16 * g);
      c0 = wmma_bf16(qf[st], b0, c0);
      c1 = wmma_bf16(qf[st], b1, c1);
    }

    // online softmax update (rows live in 16-lane groups per the D layout)
#pragma unroll
    for (int r = 0; r < 8; ++r) {
      const int s = r + 8 * g;
      const int ka = kv0 + ln, kb2 = kv0 + 16 + ln;
      float ca = c0[r] * scale, cb = c1[r] * scale;
      if (ka >= KV_ || mask[((size_t)b * S_ + s) * KV_ + ka] == 0) ca = -1e30f;
      if (kb2 >= KV_ || mask[((size_t)b * S_ + s) * KV_ + kb2] == 0) cb = -1e30f;
      float rm = fmaxf(ca, cb);
      rm = fmaxf(rm, __shfl_xor(rm, 1));
      rm = fmaxf(rm, __shfl_xor(rm, 2));
      rm = fmaxf(rm, __shfl_xor(rm, 4));
      rm = fmaxf(rm, __shfl_xor(rm, 8));
      const float mn = fmaxf(m_run[r], rm);
      const float alpha = __expf(m_run[r] - mn);
      const float p0 = __expf(ca - mn);
      const float p1 = __expf(cb - mn);
      l_run[r] = l_run[r] * alpha + p0 + p1;
      m_run[r] = mn;
#pragma unroll
      for (int j = 0; j < 8; ++j) acc[j][r] *= alpha;
      sP[s * 32 + ln]      = (__bf16)p0;   // P in [s][kv_local] for A-frag reload
      sP[s * 32 + 16 + ln] = (__bf16)p1;
    }

    // ensure async V staging landed in LDS
    wait_async0();

    // A-frag for P (16x32): lane reads its K-slices from sP row (lane&15)
    Frag16 af;
    af.q[0] = *(const v4u*)&sP[ln * 32 + 16 * g];
    af.q[1] = *(const v4u*)&sP[ln * 32 + 16 * g + 8];

    // PV: 8 d-chunks of 16; B-frag gathered column-wise from staged sV
#pragma unroll
    for (int j = 0; j < 8; ++j) {
      float t[16];
#pragma unroll
      for (int i = 0; i < 16; ++i) t[i] = sV[(16 * g + i) * 132 + j * 16 + ln];
      v16bf bv = pack16(t);
      acc[j] = wmma_bf16(af.v, bv, acc[j]);
    }
  }

  // finalize: full row sums, normalize, write [B,S,H*D]
#pragma unroll
  for (int r = 0; r < 8; ++r) {
    float l = l_run[r];
    l += __shfl_xor(l, 1);
    l += __shfl_xor(l, 2);
    l += __shfl_xor(l, 4);
    l += __shfl_xor(l, 8);
    const float inv = 1.0f / l;
    const int s = r + 8 * g;
    float* orow = O + (((size_t)b * S_ + s) * H_ + h) * D_;
#pragma unroll
    for (int j = 0; j < 8; ++j) orow[j * 16 + ln] = acc[j][r] * inv;
  }
}

// ---------------------------------------------------------------------------
extern "C" void kernel_launch(void* const* d_in, const int* in_sizes, int n_in,
                              void* d_out, int out_size, void* d_ws, size_t ws_size,
                              hipStream_t stream) {
  const float* hidden = (const float*)d_in[0];
  const int*   amask  = (const int*)d_in[1];
  const float* kcache = (const float*)d_in[2];
  const float* vcache = (const float*)d_in[3];
  const float* Wq     = (const float*)d_in[4];
  const float* Wk     = (const float*)d_in[5];
  const float* Wv     = (const float*)d_in[6];
  const float* Wo     = (const float*)d_in[7];

  float* out_o = (float*)d_out;                       // [B,S,4096]
  float* out_k = out_o + (size_t)B_ * S_ * INDIM;     // [B,H,KV,D]
  float* out_v = out_k + (size_t)B_ * H_ * KV_ * D_;  // [B,H,KV,D]

  float* q_ws = (float*)d_ws;                         // [B,H,S,D]   (2 MB)
  float* a_ws = q_ws + (size_t)B_ * H_ * S_ * D_;     // [B,S,H*D]   (2 MB)

  const dim3 blk(32);
  const dim3 grd(INDIM / 64, B_);

  // Q/K/V projections (K/V new rows land directly in the concat outputs)
  proj_gemm<<<grd, blk, 0, stream>>>(hidden, Wq, q_ws, 0);
  proj_gemm<<<grd, blk, 0, stream>>>(hidden, Wk, out_k, 1);
  proj_gemm<<<grd, blk, 0, stream>>>(hidden, Wv, out_v, 2);

  // stream the 1.07 GB of cache into the concat outputs
  const long nvec = (long)B_ * H_ * PAST_ * D_ / 4;
  copy_cache<<<4096, 256, 0, stream>>>(kcache, out_k, nvec);
  copy_cache<<<4096, 256, 0, stream>>>(vcache, out_v, nvec);

  // flash attention over the concatenated K/V
  flash_attn<<<dim3(B_ * H_), blk, 0, stream>>>(q_ws, out_k, out_v, amask, a_ws);

  // output projection
  proj_gemm<<<grd, blk, 0, stream>>>(a_ws, Wo, out_o, 3);
}
